// AttnDecoderRNN_84696755077218
// MI455X (gfx1250) — compile-verified, matrix operand order checked by reading, not run on hardware
//
#include <hip/hip_runtime.h>
#include <math.h>

#define Bq 256
#define Hq 1024
#define Lq 400
#define LPAD 416          // Lq padded to a multiple of 32 for the guard-free GEMM
#define Vq 50257

typedef __attribute__((ext_vector_type(16))) __bf16        v16bf;
typedef __attribute__((ext_vector_type(8)))  unsigned int  v8u;
typedef __attribute__((ext_vector_type(8)))  float         v8f;

union BFPack { v8u u; v16bf bf; };

// Pack two fp32 into one dword of bf16 pairs via a single v_perm_b32.
__device__ __forceinline__ unsigned int pk2(float lo, float hi) {
  return __builtin_amdgcn_perm(__float_as_uint(hi), __float_as_uint(lo), 0x07060302u);
}

__device__ __forceinline__ void pack16(BFPack& p, const float4& a, const float4& b,
                                       const float4& c, const float4& d) {
  p.u[0] = pk2(a.x, a.y); p.u[1] = pk2(a.z, a.w);
  p.u[2] = pk2(b.x, b.y); p.u[3] = pk2(b.z, b.w);
  p.u[4] = pk2(c.x, c.y); p.u[5] = pk2(c.z, c.w);
  p.u[6] = pk2(d.x, d.y); p.u[7] = pk2(d.z, d.w);
}

// Per-K-step schedule: [20 VMEM loads] -> [40 v_perm packs] -> [4 v_wmma].
__device__ __forceinline__ void sched_step() {
#if __has_builtin(__builtin_amdgcn_sched_group_barrier)
  __builtin_amdgcn_sched_group_barrier(0x020, 20, 0); // VMEM read x20
  __builtin_amdgcn_sched_group_barrier(0x002, 40, 0); // VALU x40 (perm packs)
  __builtin_amdgcn_sched_group_barrier(0x008, 4, 0);  // MFMA/WMMA x4
#endif
}

// ---------------------------------------------------------------------------
// GEMM: C[m,n] = act( sum_k A[m,k] * W[n,k] + bias[n] ),  W row-major [N,K].
// A columns [0,K0) from A0 (stride lda0), [K0,K) from A1 (stride lda1).
// Requirements: K % 32 == 0, K0 % 32 == 0, M % 64 == 0, rows 16B-aligned.
// One wave computes a 64x16 tile. waves_per_eu(2) raises the VGPR budget so
// all 20 b128 loads per step stay in distinct registers (full MLP).
// ---------------------------------------------------------------------------
__global__ __launch_bounds__(32)
__attribute__((amdgpu_waves_per_eu(2)))
void gemm_wmma_bf16(const float* __restrict__ A0, const float* __restrict__ A1,
                    int K0, int lda0, int lda1,
                    const float* __restrict__ W, int ldw,
                    const float* __restrict__ bias,
                    float* __restrict__ C, int ldc,
                    int N, int K, int relu)
{
  const int lane = threadIdx.x & 31;
  const int grp8 = (lane >> 4) * 8;   // +8 K-offset for lanes 16..31
  const int mr   = lane & 15;
  const int m0   = blockIdx.x * 64;   // 4 M-tiles per wave
  const int n0   = blockIdx.y * 16;
  const int n    = n0 + mr;
  const bool nok = (n < N);
  const int  nl  = nok ? n : (N - 1); // clamped load row; store is guarded
  const float* wrow = W + (size_t)nl * (size_t)ldw + grp8;

  v8f acc[4];
#pragma unroll
  for (int t = 0; t < 4; ++t) acc[t] = (v8f){0.f,0.f,0.f,0.f,0.f,0.f,0.f,0.f};

#pragma unroll
  for (int phase = 0; phase < 2; ++phase) {
    const int kstart = phase ? K0 : 0;
    const int kend   = phase ? K  : K0;
    if (kstart >= kend) continue;
    const float* Ap  = phase ? A1 : A0;
    const size_t lda = (size_t)(phase ? lda1 : lda0);

    const float* wp = wrow + kstart;
    const float* p0 = Ap + (size_t)(m0 + mr) * lda + grp8;
    const float* p1 = p0 + 16 * lda;
    const float* p2 = p0 + 32 * lda;
    const float* p3 = p0 + 48 * lda;

    for (int kb = kstart; kb < kend; kb += 32) {
      // ---- issue ALL 20 b128 loads into distinct locals (one clause) ----
      const float4 w0  = *(const float4*)(wp);
      const float4 w1  = *(const float4*)(wp + 4);
      const float4 w2  = *(const float4*)(wp + 16);
      const float4 w3  = *(const float4*)(wp + 20);
      const float4 x00 = *(const float4*)(p0);
      const float4 x01 = *(const float4*)(p0 + 4);
      const float4 x02 = *(const float4*)(p0 + 16);
      const float4 x03 = *(const float4*)(p0 + 20);
      const float4 x10 = *(const float4*)(p1);
      const float4 x11 = *(const float4*)(p1 + 4);
      const float4 x12 = *(const float4*)(p1 + 16);
      const float4 x13 = *(const float4*)(p1 + 20);
      const float4 x20 = *(const float4*)(p2);
      const float4 x21 = *(const float4*)(p2 + 4);
      const float4 x22 = *(const float4*)(p2 + 16);
      const float4 x23 = *(const float4*)(p2 + 20);
      const float4 x30 = *(const float4*)(p3);
      const float4 x31 = *(const float4*)(p3 + 4);
      const float4 x32 = *(const float4*)(p3 + 16);
      const float4 x33 = *(const float4*)(p3 + 20);
      wp += 32; p0 += 32; p1 += 32; p2 += 32; p3 += 32;

      // ---- pack phase (v_perm_b32 only) ----
      BFPack pb, pa0, pa1, pa2, pa3;
      pack16(pb,  w0,  w1,  w2,  w3);
      pack16(pa0, x00, x01, x02, x03);
      pack16(pa1, x10, x11, x12, x13);
      pack16(pa2, x20, x21, x22, x23);
      pack16(pa3, x30, x31, x32, x33);

      // ---- 4 WMMAs sharing the B operand ----
      acc[0] = __builtin_amdgcn_wmma_f32_16x16x32_bf16(
                 false, pa0.bf, false, pb.bf, (short)0, acc[0], false, false);
      acc[1] = __builtin_amdgcn_wmma_f32_16x16x32_bf16(
                 false, pa1.bf, false, pb.bf, (short)0, acc[1], false, false);
      acc[2] = __builtin_amdgcn_wmma_f32_16x16x32_bf16(
                 false, pa2.bf, false, pb.bf, (short)0, acc[2], false, false);
      acc[3] = __builtin_amdgcn_wmma_f32_16x16x32_bf16(
                 false, pa3.bf, false, pb.bf, (short)0, acc[3], false, false);

      sched_step();  // enforce [VMEM x20][VALU x40][WMMA x4] pipeline shape
    }
  }

  // Epilogue: lane stores column n, rows m0 + t*16 + r + grp8 (C/D layout).
  if (nok) {
    const float bn = bias ? bias[n] : 0.f;
#pragma unroll
    for (int t = 0; t < 4; ++t) {
#pragma unroll
      for (int r = 0; r < 8; ++r) {
        const int mo = m0 + t*16 + r + grp8;
        float vv = acc[t][r] + bn;
        if (relu) vv = fmaxf(vv, 0.f);
        C[(size_t)mo * (size_t)ldc + n] = vv;
      }
    }
  }
}

// ---------------------------------------------------------------------------
// Row softmax: in-place on X[:,0:ncols], plus a zero-padded copy into
// Y[:,0:npad] so the downstream GEMM can run the guard-free 32-wide K loop.
// ---------------------------------------------------------------------------
__global__ void softmax_rows_pad(float* __restrict__ X, int ncols,
                                 float* __restrict__ Y, int npad) {
  __shared__ float red[256];
  float* x = X + (size_t)blockIdx.x * (size_t)ncols;
  float* y = Y + (size_t)blockIdx.x * (size_t)npad;
  const int t = threadIdx.x;
  float mx = -INFINITY;
  for (int i = t; i < ncols; i += 256) mx = fmaxf(mx, x[i]);
  red[t] = mx; __syncthreads();
  for (int s = 128; s > 0; s >>= 1) { if (t < s) red[t] = fmaxf(red[t], red[t+s]); __syncthreads(); }
  mx = red[0]; __syncthreads();
  float sum = 0.f;
  for (int i = t; i < ncols; i += 256) { float e = __expf(x[i] - mx); x[i] = e; sum += e; }
  red[t] = sum; __syncthreads();
  for (int s = 128; s > 0; s >>= 1) { if (t < s) red[t] += red[t+s]; __syncthreads(); }
  const float inv = 1.f / red[0];
  for (int i = t; i < npad; i += 256) {
    if (i < ncols) { const float v = x[i] * inv; x[i] = v; y[i] = v; }
    else y[i] = 0.f;
  }
}

// encoder_states [Lq,Hq] -> encT [Hq,LPAD] (transposed, zero-padded K-tail)
__global__ void transpose_pad(const float* __restrict__ src, float* __restrict__ dst) {
  const int idx = blockIdx.x * blockDim.x + threadIdx.x;
  if (idx >= Hq * LPAD) return;
  const int h = idx / LPAD, l = idx % LPAD;
  dst[idx] = (l < Lq) ? src[(size_t)l * Hq + h] : 0.f;
}

// ---------------------------------------------------------------------------
// Row log-softmax in place (one block per row).
// ---------------------------------------------------------------------------
__global__ void log_softmax_rows(float* __restrict__ X, int ncols) {
  __shared__ float red[256];
  float* x = X + (size_t)blockIdx.x * (size_t)ncols;
  const int t = threadIdx.x;
  float mx = -INFINITY;
  for (int i = t; i < ncols; i += 256) mx = fmaxf(mx, x[i]);
  red[t] = mx; __syncthreads();
  for (int s = 128; s > 0; s >>= 1) { if (t < s) red[t] = fmaxf(red[t], red[t+s]); __syncthreads(); }
  mx = red[0]; __syncthreads();
  float sum = 0.f;
  for (int i = t; i < ncols; i += 256) sum += __expf(x[i] - mx);
  red[t] = sum; __syncthreads();
  for (int s = 128; s > 0; s >>= 1) { if (t < s) red[t] += red[t+s]; __syncthreads(); }
  const float lse = mx + logf(red[0]);
  for (int i = t; i < ncols; i += 256) x[i] -= lse;
}

// ---------------------------------------------------------------------------
// GRU gates (PyTorch order r,z,n): h_new = (1-z)*tanh(i_n + r*h_n) + z*h
// ---------------------------------------------------------------------------
__global__ void gru_gate(const float* __restrict__ gi, const float* __restrict__ gh,
                         const float* __restrict__ hprev, float* __restrict__ hnew,
                         int Bv, int Hv) {
  const int idx = blockIdx.x * blockDim.x + threadIdx.x;
  if (idx >= Bv * Hv) return;
  const int b = idx / Hv, j = idx % Hv;
  const float* gib = gi + (size_t)b * (size_t)(3 * Hv);
  const float* ghb = gh + (size_t)b * (size_t)(3 * Hv);
  const float r = 1.f / (1.f + __expf(-(gib[j]      + ghb[j])));
  const float z = 1.f / (1.f + __expf(-(gib[Hv + j] + ghb[Hv + j])));
  const float nn = tanhf(gib[2*Hv + j] + r * ghb[2*Hv + j]);
  hnew[idx] = (1.f - z) * nn + z * hprev[idx];
}

extern "C" void kernel_launch(void* const* d_in, const int* in_sizes, int n_in,
                              void* d_out, int out_size, void* d_ws, size_t ws_size,
                              hipStream_t stream) {
  (void)in_sizes; (void)n_in; (void)out_size; (void)ws_size;
  const float* embedded = (const float*)d_in[0];   // [B,H]
  const float* hidden   = (const float*)d_in[1];   // [B,H]
  const float* enc      = (const float*)d_in[2];   // [L,H]
  const float* attn_W   = (const float*)d_in[3];   // [L,2H]
  const float* attn_b   = (const float*)d_in[4];   // [L]
  const float* comb_W   = (const float*)d_in[5];   // [H,2H]
  const float* comb_b   = (const float*)d_in[6];   // [H]
  const float* gru_Wih  = (const float*)d_in[7];   // [3H,H]
  const float* gru_Whh  = (const float*)d_in[8];   // [3H,H]
  const float* gru_bih  = (const float*)d_in[9];   // [3H]
  const float* gru_bhh  = (const float*)d_in[10];  // [3H]
  const float* out_W    = (const float*)d_in[11];  // [V,H]
  const float* out_b    = (const float*)d_in[12];  // [V]

  float* logp  = (float*)d_out;                    // [B,V]
  float* hnew  = logp + (size_t)Bq * Vq;           // [B,H]
  float* attnw = hnew + (size_t)Bq * Hq;           // [B,L]

  float* ws      = (float*)d_ws;
  float* ws_o    = ws;                             // [B,H]
  float* ws_appl = ws_o    + (size_t)Bq * Hq;      // [B,H]
  float* ws_gi   = ws_appl + (size_t)Bq * Hq;      // [B,3H]
  float* ws_gh   = ws_gi   + (size_t)Bq * 3 * Hq;  // [B,3H]
  float* ws_attn = ws_gh   + (size_t)Bq * 3 * Hq;  // [B,LPAD]
  float* ws_encT = ws_attn + (size_t)Bq * LPAD;    // [H,LPAD]

  const dim3 blk(32);

  // 0) one-time transpose+pad of encoder states: [L,H] -> [H,LPAD]
  transpose_pad<<<(Hq * LPAD + 255) / 256, 256, 0, stream>>>(enc, ws_encT);

  // 1) attn scores = [x||h] @ attn_W^T + attn_b  -> attnw (d_out region)
  gemm_wmma_bf16<<<dim3(Bq/64, Lq/16), blk, 0, stream>>>(
      embedded, hidden, /*K0=*/Hq, Hq, Hq,
      attn_W, /*ldw=*/2*Hq, attn_b,
      attnw, /*ldc=*/Lq, /*N=*/Lq, /*K=*/2*Hq, /*relu=*/0);

  // 2) softmax over L, plus zero-padded copy for the next GEMM
  softmax_rows_pad<<<Bq, 256, 0, stream>>>(attnw, Lq, ws_attn, LPAD);

  // 3) attn_applied = attn_weights @ enc   (K padded to 416 = 13*32)
  gemm_wmma_bf16<<<dim3(Bq/64, Hq/16), blk, 0, stream>>>(
      ws_attn, ws_attn, /*K0=*/LPAD, LPAD, LPAD,
      ws_encT, /*ldw=*/LPAD, nullptr,
      ws_appl, /*ldc=*/Hq, /*N=*/Hq, /*K=*/LPAD, /*relu=*/0);

  // 4) o = relu([x||attn_applied] @ comb_W^T + comb_b)
  gemm_wmma_bf16<<<dim3(Bq/64, Hq/16), blk, 0, stream>>>(
      embedded, ws_appl, /*K0=*/Hq, Hq, Hq,
      comb_W, /*ldw=*/2*Hq, comb_b,
      ws_o, /*ldc=*/Hq, /*N=*/Hq, /*K=*/2*Hq, /*relu=*/1);

  // 5) gi = o @ gru_Wih^T + bih ; gh = h @ gru_Whh^T + bhh
  gemm_wmma_bf16<<<dim3(Bq/64, (3*Hq)/16), blk, 0, stream>>>(
      ws_o, ws_o, /*K0=*/Hq, Hq, Hq,
      gru_Wih, Hq, gru_bih, ws_gi, 3*Hq, 3*Hq, Hq, 0);
  gemm_wmma_bf16<<<dim3(Bq/64, (3*Hq)/16), blk, 0, stream>>>(
      hidden, hidden, /*K0=*/Hq, Hq, Hq,
      gru_Whh, Hq, gru_bhh, ws_gh, 3*Hq, 3*Hq, Hq, 0);

  // 6) GRU gate -> h_new
  gru_gate<<<(Bq*Hq + 255)/256, 256, 0, stream>>>(ws_gi, ws_gh, hidden, hnew, Bq, Hq);

  // 7) logits = h_new @ out_W^T + out_b -> logp (206 MB weight stream; the
  //    dominant, bandwidth-bound GEMM: ~8.8 us at 23.3 TB/s)
  gemm_wmma_bf16<<<dim3(Bq/64, (Vq + 15)/16), blk, 0, stream>>>(
      hnew, hnew, /*K0=*/Hq, Hq, Hq,
      out_W, Hq, out_b, logp, Vq, Vq, Hq, 0);

  // 8) log-softmax over V
  log_softmax_rows<<<Bq, 256, 0, stream>>>(logp, Vq);
}
